// LongformerEncoder_55164559950295
// MI455X (gfx1250) — compile-verified, matrix operand order checked by reading, not run on hardware
//
#include <hip/hip_runtime.h>
#include <hip/hip_bf16.h>

// ---------------- problem constants ----------------
#define BB 2
#define SS 2048
#define HH 512
#define NHH 8
#define HDD 64
#define LL 4
#define INTERR 2048
#define W1W 32
#define MROWS (BB * SS)   // 4096

// ---------------- WMMA types ----------------
typedef __attribute__((ext_vector_type(16))) __bf16 v16bf;
typedef __attribute__((ext_vector_type(8)))  float  v8f;

struct U4 { unsigned int x, y, z, w; };
union Frag16 { v16bf v; U4 q[2]; unsigned short u[16]; };

__device__ __forceinline__ unsigned short f32_to_bf16(float f) {
    unsigned int u = __float_as_uint(f);
    unsigned int r = (u + 0x7FFFu + ((u >> 16) & 1u)) >> 16;
    return (unsigned short)r;
}

// =====================================================================
// Generic bf16-WMMA GEMM: C[M,N] = act(A[M,K] @ W[K,N] + bias + residual)*scale
// ABF16 / ACT are compile-time (branch-free staging + epilogue).
// Block: 256 thr (8 waves). Tile: 128(M) x 64(N), K step 32.
// Wave w computes rows [w*16, w*16+16) x all 64 cols (4 WMMAs per K step).
// =====================================================================
#define GK 32
#define GMT 128
#define GNT 64

template <int ABF16, int ACT /*0 none,1 gelu,2 tanh*/>
__global__ __launch_bounds__(256)
void lf_gemm_wmma(const void* __restrict__ Ap,
                  const float* __restrict__ Wt, const float* __restrict__ bias,
                  const float* __restrict__ residual,
                  float* __restrict__ outF, unsigned short* __restrict__ outB,
                  int M, int K, int N, float scale) {
    __shared__ __align__(16) unsigned short Asm[GMT][GK];      // 8 KB
    __shared__ __align__(16) unsigned short Bsm[GNT][GK + 16]; // 6 KB (row=96B, 16B-aligned)

    const int tid  = threadIdx.x;
    const int lane = tid & 31;
    const int wave = tid >> 5;
    const int hi   = lane >> 4;        // 0: K {0..7,16..23}; 1: K {8..15,24..31}
    const int l16  = lane & 15;
    const int mBase = blockIdx.y * GMT;
    const int nBase = blockIdx.x * GNT;

    const float* Af = (const float*)Ap;
    const unsigned short* Ab = (const unsigned short*)Ap;

    v8f zero = {0.f,0.f,0.f,0.f,0.f,0.f,0.f,0.f};
    v8f acc[4]; acc[0]=zero; acc[1]=zero; acc[2]=zero; acc[3]=zero;

    for (int k0 = 0; k0 < K; k0 += GK) {
        // ---- stage A tile (128 x 32), branch-free ----
        #pragma unroll
        for (int i = 0; i < (GMT * GK) / 256; i++) {
            const int e = tid + i * 256;
            const int r = e >> 5, c = e & 31;
            const size_t g = (size_t)(mBase + r) * K + k0 + c;
            Asm[r][c] = ABF16 ? Ab[g] : f32_to_bf16(Af[g]);
        }
        // ---- stage W tile transposed (row = n, col = k), coalesced over n ----
        #pragma unroll
        for (int i = 0; i < (GNT * GK) / 256; i++) {
            const int e = tid + i * 256;
            const int r = e & 63, c = e >> 6;
            Bsm[r][c] = f32_to_bf16(Wt[(size_t)(k0 + c) * N + nBase + r]);
        }
        if (k0 + GK < K) { // prefetch next weight tile (global_prefetch_b8)
            __builtin_prefetch(&Wt[(size_t)(k0 + GK) * N + nBase + l16], 0, 1);
        }
        __syncthreads();

        Frag16 a;
        const int m = wave * 16 + l16;
        a.q[0] = *(const U4*)&Asm[m][hi * 8];
        a.q[1] = *(const U4*)&Asm[m][16 + hi * 8];
        #pragma unroll
        for (int nt = 0; nt < 4; nt++) {
            Frag16 b;
            const int n = nt * 16 + l16;
            b.q[0] = *(const U4*)&Bsm[n][hi * 8];
            b.q[1] = *(const U4*)&Bsm[n][16 + hi * 8];
            acc[nt] = __builtin_amdgcn_wmma_f32_16x16x32_bf16(
                false, a.v, false, b.v, (short)0, acc[nt], false, false);
        }
        __syncthreads();
    }

    // ---- epilogue: C layout lane=N (l16), VGPR i -> M = hi*8 + i ----
    #pragma unroll
    for (int nt = 0; nt < 4; nt++) {
        const int n = nBase + nt * 16 + l16;
        const float bv = bias ? bias[n] : 0.0f;
        #pragma unroll
        for (int i = 0; i < 8; i++) {
            const int m = mBase + wave * 16 + hi * 8 + i;
            float v = acc[nt][i] + bv;
            if (residual) v += residual[(size_t)m * N + n];
            if (ACT == 1) v = 0.5f * v * (1.0f + erff(v * 0.70710678118f)); // exact GELU
            else if (ACT == 2) v = tanhf(v);
            v *= scale;
            const size_t off = (size_t)m * N + n;
            if (outF) outF[off] = v;
            if (outB) outB[off] = f32_to_bf16(v);
        }
    }
}

// =====================================================================
// Sliding-window attention, one wave per (b, h, 16-query tile).
// Q/K/V bf16 (B,S,NH,HD), Q pre-scaled. ctx f32 (B,S,H) with H = NH*HD.
// =====================================================================
__global__ __launch_bounds__(32)
void lf_attn_window(const unsigned short* __restrict__ Q,
                    const unsigned short* __restrict__ Kb,
                    const unsigned short* __restrict__ Vb,
                    const float* __restrict__ Mask,
                    float* __restrict__ ctx) {
    const int lane = threadIdx.x;
    const int l16  = lane & 15;
    const int hi   = lane >> 4;
    const int s0   = blockIdx.x * 16;
    const int h    = blockIdx.y;
    const int b    = blockIdx.z;

    __shared__ __align__(16) unsigned short Plds[16][96]; // P tile, zero-padded K to 96

    for (int i = lane; i < 16 * 96; i += 32) ((unsigned short*)Plds)[i] = 0;

    // ---- load Q fragments (A layout), rows s0..s0+15, K=64 in two chunks ----
    Frag16 qa[2];
    {
        const int s = s0 + l16;
        const size_t qrow = ((size_t)(b * SS + s) * NHH + h) * HDD;
        #pragma unroll
        for (int c = 0; c < 2; c++) {
            qa[c].q[0] = *(const U4*)&Q[qrow + c * 32 + hi * 8];
            qa[c].q[1] = *(const U4*)&Q[qrow + c * 32 + 16 + hi * 8];
        }
    }

    // ---- scores: 5 key tiles of 16 spanning offsets [-32, +47] ----
    v8f zero = {0.f,0.f,0.f,0.f,0.f,0.f,0.f,0.f};
    v8f sc[5];
    #pragma unroll
    for (int kt = 0; kt < 5; kt++) {
        const int kbase = s0 - 32 + kt * 16;
        int key = kbase + l16;
        int keyc = min(max(key, 0), SS - 1);
        const size_t krow = ((size_t)(b * SS + keyc) * NHH + h) * HDD;
        v8f a = zero;
        #pragma unroll
        for (int c = 0; c < 2; c++) {
            Frag16 kf;
            kf.q[0] = *(const U4*)&Kb[krow + c * 32 + hi * 8];
            kf.q[1] = *(const U4*)&Kb[krow + c * 32 + 16 + hi * 8];
            a = __builtin_amdgcn_wmma_f32_16x16x32_bf16(
                false, qa[c].v, false, kf.v, (short)0, a, false, false);
        }
        sc[kt] = a;
    }

    // ---- mask + key bias, then row softmax (rows split: lanes<16 hold M=0..7) ----
    #pragma unroll
    for (int kt = 0; kt < 5; kt++) {
        const int key = s0 - 32 + kt * 16 + l16;
        #pragma unroll
        for (int i = 0; i < 8; i++) {
            const int s = s0 + hi * 8 + i;
            bool valid = (key >= s - W1W) && (key <= s + W1W) && (key >= 0) && (key < SS);
            if (valid) {
                float ext = (1.0f - Mask[(size_t)b * SS + key]) * -10000.0f;
                sc[kt][i] += ext;
            } else {
                sc[kt][i] = -1.0e30f;
            }
        }
    }
    float mrow[8], srow[8];
    #pragma unroll
    for (int i = 0; i < 8; i++) {
        float mv = sc[0][i];
        #pragma unroll
        for (int kt = 1; kt < 5; kt++) mv = fmaxf(mv, sc[kt][i]);
        #pragma unroll
        for (int off = 1; off < 16; off <<= 1) mv = fmaxf(mv, __shfl_xor(mv, off, 32));
        mrow[i] = mv;
    }
    #pragma unroll
    for (int i = 0; i < 8; i++) {
        float sv = 0.0f;
        #pragma unroll
        for (int kt = 0; kt < 5; kt++) { sc[kt][i] = __expf(sc[kt][i] - mrow[i]); sv += sc[kt][i]; }
        #pragma unroll
        for (int off = 1; off < 16; off <<= 1) sv += __shfl_xor(sv, off, 32);
        srow[i] = sv;
    }
    #pragma unroll
    for (int i = 0; i < 8; i++) {
        const int s = s0 + hi * 8 + i;
        const bool qMasked = ((1.0f - Mask[(size_t)b * SS + s]) * -10000.0f) < 0.0f;
        const float inv = qMasked ? 0.0f : 1.0f / srow[i];
        #pragma unroll
        for (int kt = 0; kt < 5; kt++) {
            Plds[hi * 8 + i][kt * 16 + l16] = f32_to_bf16(sc[kt][i] * inv);
        }
    }
    // single wave: make P stores visible before the A-layout re-reads
    asm volatile("s_wait_dscnt 0" ::: "memory");

    // ---- ctx = P(16x96, zero-padded) @ V ----
    v8f cacc[4]; cacc[0]=zero; cacc[1]=zero; cacc[2]=zero; cacc[3]=zero;
    #pragma unroll
    for (int kc = 0; kc < 3; kc++) {
        Frag16 pa;
        pa.q[0] = *(const U4*)&Plds[l16][kc * 32 + hi * 8];
        pa.q[1] = *(const U4*)&Plds[l16][kc * 32 + 16 + hi * 8];
        #pragma unroll
        for (int nt = 0; nt < 4; nt++) {
            const int d = nt * 16 + l16;
            Frag16 vf;
            #pragma unroll
            for (int e = 0; e < 16; e++) {
                const int kk = kc * 32 + hi * 8 + ((e < 8) ? e : (16 + e - 8));
                int key = s0 - 32 + kk;
                int keyc = min(max(key, 0), SS - 1);
                vf.u[e] = Vb[((size_t)(b * SS + keyc) * NHH + h) * HDD + d];
            }
            cacc[nt] = __builtin_amdgcn_wmma_f32_16x16x32_bf16(
                false, pa.v, false, vf.v, (short)0, cacc[nt], false, false);
        }
    }

    // ---- store ctx f32, (B,S,H) with head-major columns ----
    #pragma unroll
    for (int nt = 0; nt < 4; nt++) {
        const int d = nt * 16 + l16;
        #pragma unroll
        for (int i = 0; i < 8; i++) {
            const int s = s0 + hi * 8 + i;
            ctx[((size_t)(b * SS + s) * NHH + h) * HDD + d] = cacc[nt][i];
        }
    }
}

// =====================================================================
// x[b,s,:] = input + pos  (flat over B*S*H)
// =====================================================================
__global__ __launch_bounds__(256)
void lf_add_pos(const float* __restrict__ emb, const float* __restrict__ pos,
                float* __restrict__ out) {
    size_t i = (size_t)blockIdx.x * 256 + threadIdx.x;
    out[i] = emb[i] + pos[i % ((size_t)SS * HH)];
}

// =====================================================================
// LayerNorm over H=512, one row per block (256 thr, 2 elems each)
// =====================================================================
__global__ __launch_bounds__(256)
void lf_layernorm(const float* __restrict__ in, const float* __restrict__ g,
                  const float* __restrict__ bta, float* __restrict__ outF) {
    const int row = blockIdx.x, tid = threadIdx.x;
    const float v0 = in[(size_t)row * HH + tid];
    const float v1 = in[(size_t)row * HH + tid + 256];
    float s = v0 + v1, s2 = v0 * v0 + v1 * v1;
    #pragma unroll
    for (int off = 1; off < 32; off <<= 1) { s += __shfl_xor(s, off); s2 += __shfl_xor(s2, off); }
    __shared__ float r1[8], r2[8];
    const int wave = tid >> 5, lane = tid & 31;
    if (lane == 0) { r1[wave] = s; r2[wave] = s2; }
    __syncthreads();
    if (wave == 0) {
        float a = (lane < 8) ? r1[lane] : 0.f, c = (lane < 8) ? r2[lane] : 0.f;
        #pragma unroll
        for (int off = 1; off < 8; off <<= 1) { a += __shfl_xor(a, off); c += __shfl_xor(c, off); }
        if (lane == 0) { r1[0] = a; r2[0] = c; }
    }
    __syncthreads();
    const float mean = r1[0] * (1.0f / HH);
    const float var  = r2[0] * (1.0f / HH) - mean * mean;
    const float inv  = rsqrtf(var + 1e-12f);
    outF[(size_t)row * HH + tid]       = (v0 - mean) * inv * g[tid] + bta[tid];
    outF[(size_t)row * HH + tid + 256] = (v1 - mean) * inv * g[tid + 256] + bta[tid + 256];
}

// =====================================================================
// alpha_raw[row] = exp(dot(e[row,:], W2) + b2) * mask[row]
// =====================================================================
__global__ __launch_bounds__(256)
void lf_pool_logits(const float* __restrict__ e, const float* __restrict__ W2,
                    const float* __restrict__ b2, const float* __restrict__ Mask,
                    float* __restrict__ alpha) {
    const int row = blockIdx.x, tid = threadIdx.x;
    float s = e[(size_t)row * HH + tid] * W2[tid]
            + e[(size_t)row * HH + tid + 256] * W2[tid + 256];
    #pragma unroll
    for (int off = 1; off < 32; off <<= 1) s += __shfl_xor(s, off);
    __shared__ float r1[8];
    const int wave = tid >> 5, lane = tid & 31;
    if (lane == 0) r1[wave] = s;
    __syncthreads();
    if (tid == 0) {
        float a = 0.f;
        #pragma unroll
        for (int w = 0; w < 8; w++) a += r1[w];
        alpha[row] = __expf(a + b2[0]) * Mask[row];
    }
}

__global__ __launch_bounds__(256)
void lf_pool_sums(const float* __restrict__ alpha, float* __restrict__ sums) {
    const int b = blockIdx.x, tid = threadIdx.x;
    float s = 0.f;
    for (int i = tid; i < SS; i += 256) s += alpha[(size_t)b * SS + i];
    #pragma unroll
    for (int off = 1; off < 32; off <<= 1) s += __shfl_xor(s, off);
    __shared__ float r1[8];
    const int wave = tid >> 5, lane = tid & 31;
    if (lane == 0) r1[wave] = s;
    __syncthreads();
    if (tid == 0) {
        float a = 0.f;
        #pragma unroll
        for (int w = 0; w < 8; w++) a += r1[w];
        sums[b] = a;
    }
}

__global__ __launch_bounds__(512)
void lf_pool_out(const float* __restrict__ x, const float* __restrict__ alpha,
                 const float* __restrict__ sums, float* __restrict__ out) {
    const int b = blockIdx.x, hcol = threadIdx.x;
    const float inv = 1.0f / (sums[b] + 1e-8f);
    float acc = 0.f;
    for (int s = 0; s < SS; s++)
        acc += x[((size_t)b * SS + s) * HH + hcol] * alpha[(size_t)b * SS + s];
    out[(size_t)b * HH + hcol] = acc * inv;
}

// =====================================================================
// Host side
// =====================================================================
extern "C" void kernel_launch(void* const* d_in, const int* in_sizes, int n_in,
                              void* d_out, int out_size, void* d_ws, size_t ws_size,
                              hipStream_t stream) {
    const float* input_embs = (const float*)d_in[0];
    const float* mask       = (const float*)d_in[1];
    const float* pos        = (const float*)d_in[2];
    const float* ln_emb_g   = (const float*)d_in[3];
    const float* ln_emb_b   = (const float*)d_in[4];
    const float* Wq  = (const float*)d_in[5];   const float* bq  = (const float*)d_in[6];
    const float* Wk  = (const float*)d_in[7];   const float* bk  = (const float*)d_in[8];
    const float* Wv  = (const float*)d_in[9];   const float* bv  = (const float*)d_in[10];
    const float* Wo  = (const float*)d_in[11];  const float* bo  = (const float*)d_in[12];
    const float* ln1g = (const float*)d_in[13]; const float* ln1b = (const float*)d_in[14];
    const float* Wi  = (const float*)d_in[15];  const float* bi  = (const float*)d_in[16];
    const float* Wo2 = (const float*)d_in[17];  const float* bo2 = (const float*)d_in[18];
    const float* ln2g = (const float*)d_in[19]; const float* ln2b = (const float*)d_in[20];
    const float* pW1 = (const float*)d_in[21];  const float* pb1 = (const float*)d_in[22];
    const float* pW2 = (const float*)d_in[23];  const float* pb2 = (const float*)d_in[24];

    char* ws = (char*)d_ws;
    const size_t SZF = (size_t)MROWS * HH * sizeof(float);          // 8 MB
    const size_t SZB = (size_t)MROWS * HH * sizeof(unsigned short); // 4 MB
    float*          x     = (float*)(ws);
    float*          tmp   = (float*)(ws + SZF);
    float*          attn  = (float*)(ws + 2 * SZF);
    float*          ctx   = (float*)(ws + 3 * SZF);     // reused as pooler 'e'
    unsigned short* qb    = (unsigned short*)(ws + 4 * SZF);
    unsigned short* kb    = (unsigned short*)(ws + 4 * SZF + SZB);
    unsigned short* vb    = (unsigned short*)(ws + 4 * SZF + 2 * SZB);
    unsigned short* inter = (unsigned short*)(ws + 4 * SZF + 3 * SZB);
    float*          alpha = (float*)(ws + 4 * SZF + 3 * SZB +
                                     (size_t)MROWS * INTERR * sizeof(unsigned short));
    float*          sums  = alpha + MROWS;

    dim3 blk(256);
    auto grid = [](int M, int N) { return dim3(N / GNT, M / GMT); };

    // embeddings + LN
    lf_add_pos<<<(MROWS * HH) / 256, 256, 0, stream>>>(input_embs, pos, tmp);
    lf_layernorm<<<MROWS, 256, 0, stream>>>(tmp, ln_emb_g, ln_emb_b, x);

    const float qscale = 0.125f; // 1/sqrt(HD=64)
    for (int l = 0; l < LL; l++) {
        const size_t oW = (size_t)l * HH * HH, oB = (size_t)l * HH;
        const size_t oWi = (size_t)l * HH * INTERR, oBi = (size_t)l * INTERR;
        const size_t oWo2 = (size_t)l * INTERR * HH;

        lf_gemm_wmma<0,0><<<grid(MROWS,HH), blk, 0, stream>>>(
            x, Wq + oW, bq + oB, nullptr, nullptr, qb, MROWS, HH, HH, qscale);
        lf_gemm_wmma<0,0><<<grid(MROWS,HH), blk, 0, stream>>>(
            x, Wk + oW, bk + oB, nullptr, nullptr, kb, MROWS, HH, HH, 1.0f);
        lf_gemm_wmma<0,0><<<grid(MROWS,HH), blk, 0, stream>>>(
            x, Wv + oW, bv + oB, nullptr, nullptr, vb, MROWS, HH, HH, 1.0f);

        lf_attn_window<<<dim3(SS / 16, NHH, BB), 32, 0, stream>>>(qb, kb, vb, mask, ctx);

        lf_gemm_wmma<0,0><<<grid(MROWS,HH), blk, 0, stream>>>(
            ctx, Wo + oW, bo + oB, x, tmp, nullptr, MROWS, HH, HH, 1.0f);
        lf_layernorm<<<MROWS, 256, 0, stream>>>(tmp, ln1g + oB, ln1b + oB, attn);

        lf_gemm_wmma<0,1><<<grid(MROWS,INTERR), blk, 0, stream>>>(
            attn, Wi + oWi, bi + oBi, nullptr, nullptr, inter, MROWS, HH, INTERR, 1.0f);
        lf_gemm_wmma<1,0><<<grid(MROWS,HH), blk, 0, stream>>>(
            inter, Wo2 + oWo2, bo2 + oB, attn, tmp, nullptr, MROWS, INTERR, HH, 1.0f);
        lf_layernorm<<<MROWS, 256, 0, stream>>>(tmp, ln2g + oB, ln2b + oB, x);
    }

    // pooler
    lf_gemm_wmma<0,2><<<grid(MROWS,HH), blk, 0, stream>>>(
        x, pW1, pb1, nullptr, ctx, nullptr, MROWS, HH, HH, 1.0f);
    lf_pool_logits<<<MROWS, 256, 0, stream>>>(ctx, pW2, pb2, mask, alpha);
    lf_pool_sums<<<BB, 256, 0, stream>>>(alpha, sums);
    lf_pool_out<<<BB, HH, 0, stream>>>(x, alpha, sums, (float*)d_out);
    (void)in_sizes; (void)n_in; (void)out_size; (void)ws_size;
}
